// neuralODE_49366354100337
// MI455X (gfx1250) — matrix operand order repaired and checked by reading, not compiled, not run on hardware
//
#include <hip/hip_runtime.h>
#include <math.h>

#define HDIM 64
#define WPB  8          // waves per block
#define TILE 16         // spring instances per wave
#define L_EFF 0.1f

typedef __attribute__((ext_vector_type(16))) __bf16 v16bf;
typedef __attribute__((ext_vector_type(8)))  __bf16 v8bf;
typedef __attribute__((ext_vector_type(8)))  float  v8f;

union BF16x16 { v16bf v; v8bf h[2]; __bf16 e[16]; uint4 q[2]; };

// fast, branch-free tanh: 1 - 2/(exp2(2x*log2e)+1); exact at +-inf
__device__ __forceinline__ float fast_tanh(float x) {
    const float e = __builtin_amdgcn_exp2f(x * 2.8853900817779268f); // 2*log2(e)
    return 1.0f - 2.0f * __builtin_amdgcn_rcpf(e + 1.0f);
}

// ---------------------------------------------------------------- zero ws ---
__global__ void zero_kernel(float* __restrict__ p, int n) {
    int i = blockIdx.x * blockDim.x + threadIdx.x;
    if (i < n) p[i] = 0.0f;
}

// ---------------------------------------------- pack W2 / W2^T B-fragments --
// Fragment index f = dir*8 + kstep*4 + t  (dir 0: W2, dir 1: W2^T).
// Per-lane 16 contiguous bf16 (32B) in exact B-operand register order:
//   element e -> K = kstep*32 + (lane>=16 ? 8 : 0) + (e<8 ? e : 8+e) , n = t*16 + (lane&15)
__global__ void prep_frag_kernel(const float* __restrict__ W2,
                                 __bf16* __restrict__ frag)
{
    const int tid = threadIdx.x;
    for (int pair = tid; pair < 16 * 32; pair += blockDim.x) {
        const int f    = pair >> 5;
        const int lane = pair & 31;
        const int dir   = f >> 3;
        const int kstep = (f >> 2) & 1;
        const int t     = f & 3;
        const int n     = t * 16 + (lane & 15);
        const int kbase = kstep * 32 + ((lane & 16) ? 8 : 0);
        __bf16* dst = frag + (size_t)f * 512 + lane * 16;
        #pragma unroll
        for (int e = 0; e < 16; ++e) {
            const int K = kbase + (e < 8 ? e : 8 + e);
            const float v = dir ? W2[n * HDIM + K] : W2[K * HDIM + n];
            dst[e] = (__bf16)v;
        }
    }
}

// ------------------------------------------------ fused spring-force pass ---
__global__ __launch_bounds__(256) void spring_force_kernel(
    const float* __restrict__ x, const int* __restrict__ springs,
    const float* __restrict__ W1, const float* __restrict__ b1,
    const float* __restrict__ b2, const float* __restrict__ W3,
    const __bf16* __restrict__ frag,
    float* __restrict__ f_el, int B, int S, int ndof)
{
    __shared__ __bf16 sh_h1[WPB][TILE * HDIM];   // h1 tile per wave
    __shared__ __bf16 sh_dz[WPB][TILE * HDIM];   // dz2 tile per wave
    __shared__ float  sh_g [WPB][TILE * 2];      // (g_eps, g_kappa) per row

    const int lane = threadIdx.x & 31;
    const int wid  = threadIdx.x >> 5;
    const int r    = lane & 15;                 // matrix row this lane owns
    const int hi   = (lane >= 16) ? 1 : 0;

    __bf16* h1p = sh_h1[wid];
    __bf16* dzp = sh_dz[wid];
    float*  gp  = sh_g[wid];

    const long long total = (long long)B * S;
    const long long base  = ((long long)blockIdx.x * WPB + wid) * TILE;

    // ---- gather + strain (each half-wave redundantly computes rows 0..15) --
    long long inst = base + r;
    const bool valid = inst < total;
    if (!valid) inst = total - 1;
    const int bb = (int)(inst / S);
    const int ss = (int)(inst - (long long)bb * S);
    const float* qb = x + (long long)bb * (2 * ndof);
    const int i0 = springs[ss * 3 + 0];
    const int i1 = springs[ss * 3 + 1];
    const int i2 = springs[ss * 3 + 2];

    const float ax = qb[i0*3+0], ay = qb[i0*3+1], az = qb[i0*3+2];
    const float bx = qb[i1*3+0], by = qb[i1*3+1], bz = qb[i1*3+2];
    const float cx = qb[i2*3+0], cy = qb[i2*3+1], cz = qb[i2*3+2];

    const float e1x = bx-ax, e1y = by-ay, e1z = bz-az;
    const float e2x = cx-bx, e2y = cy-by, e2z = cz-bz;
    const float n1 = sqrtf(e1x*e1x + e1y*e1y + e1z*e1z);
    const float n2 = sqrtf(e2x*e2x + e2y*e2y + e2z*e2z);
    const float dt = e1x*e2x + e1y*e2y + e1z*e2z;
    const float D  = n1 * n2 + dt;
    const float crx = e1y*e2z - e1z*e2y;
    const float cry = e1z*e2x - e1x*e2z;
    const float crz = e1x*e2y - e1y*e2x;
    const float i2D = 2.0f / D;
    const float kbx = crx * i2D, kby = cry * i2D, kbz = crz * i2D;
    const float kappa = sqrtf(kbx*kbx + kby*kby + kbz*kbz);
    const float s0 = 0.5f * (n1 * (1.0f/L_EFF) + n2 * (1.0f/L_EFF) - 2.0f);
    const float s1 = kappa;

    // ---- layer 1 (2 -> 64): lane covers 32 contiguous columns of its row ---
    {
        const int colbase = hi ? 32 : 0;
        const float* W1r0 = W1 + colbase;
        const float* W1r1 = W1 + HDIM + colbase;
        const float* b1c  = b1 + colbase;
        #pragma unroll
        for (int cc = 0; cc < 4; ++cc) {
            v8bf pack;
            #pragma unroll
            for (int e = 0; e < 8; ++e) {
                const int c = cc * 8 + e;
                const float z = fmaf(s0, W1r0[c], fmaf(s1, W1r1[c], b1c[c]));
                pack[e] = (__bf16)fast_tanh(z);
            }
            *(v8bf*)&h1p[r * HDIM + colbase + cc * 8] = pack;
        }
    }
    __syncthreads();

    // ---- A fragments of h1: two contiguous 16B runs per 16x32 fragment ----
    BF16x16 A0, A1;
    A0.h[0] = *(const v8bf*)&h1p[r * HDIM +      hi * 8];
    A0.h[1] = *(const v8bf*)&h1p[r * HDIM + 16 + hi * 8];
    A1.h[0] = *(const v8bf*)&h1p[r * HDIM + 32 + hi * 8];
    A1.h[1] = *(const v8bf*)&h1p[r * HDIM + 48 + hi * 8];

    // ---- forward WMMA: z2 = h1 @ W2  (16x64 @ 64x64) ----
    const uint4* fq = (const uint4*)frag;   // fragment f: 64 uint4, lane: 2
    v8f acc[4];
    #pragma unroll
    for (int t = 0; t < 4; ++t) {
        v8f a = {};
        BF16x16 Bf;
        const uint4* p0 = fq + (size_t)(0 * 4 + t) * 64 + lane * 2;   // kstep 0
        Bf.q[0] = p0[0]; Bf.q[1] = p0[1];
        a = __builtin_amdgcn_wmma_f32_16x16x32_bf16(false, A0.v, false, Bf.v,
                                                    (short)0, a, false, false);
        const uint4* p1 = fq + (size_t)(1 * 4 + t) * 64 + lane * 2;   // kstep 1
        Bf.q[0] = p1[0]; Bf.q[1] = p1[1];
        a = __builtin_amdgcn_wmma_f32_16x16x32_bf16(false, A1.v, false, Bf.v,
                                                    (short)0, a, false, false);
        acc[t] = a;
    }

    // ---- h2 = tanh(z2+b2);  dz2 = W3 * (1-h2^2)  (in C layout) ----
    #pragma unroll
    for (int t = 0; t < 4; ++t) {
        const int col = t * 16 + r;
        const float w3c = W3[col];
        const float b2c = b2[col];
        #pragma unroll
        for (int rr = 0; rr < 8; ++rr) {
            const int M = rr + (hi ? 8 : 0);
            const float h2 = fast_tanh(acc[t][rr] + b2c);
            dzp[M * HDIM + col] = (__bf16)(w3c * (1.0f - h2 * h2));
        }
    }
    __syncthreads();

    // ---- A fragments of dz2 ----
    A0.h[0] = *(const v8bf*)&dzp[r * HDIM +      hi * 8];
    A0.h[1] = *(const v8bf*)&dzp[r * HDIM + 16 + hi * 8];
    A1.h[0] = *(const v8bf*)&dzp[r * HDIM + 32 + hi * 8];
    A1.h[1] = *(const v8bf*)&dzp[r * HDIM + 48 + hi * 8];

    // ---- backward WMMA: dh1 = dz2 @ W2^T ----
    v8f acc2[4];
    #pragma unroll
    for (int t = 0; t < 4; ++t) {
        v8f a = {};
        BF16x16 Bf;
        const uint4* p0 = fq + (size_t)(8 + 0 * 4 + t) * 64 + lane * 2;
        Bf.q[0] = p0[0]; Bf.q[1] = p0[1];
        a = __builtin_amdgcn_wmma_f32_16x16x32_bf16(false, A0.v, false, Bf.v,
                                                    (short)0, a, false, false);
        const uint4* p1 = fq + (size_t)(8 + 1 * 4 + t) * 64 + lane * 2;
        Bf.q[0] = p1[0]; Bf.q[1] = p1[1];
        a = __builtin_amdgcn_wmma_f32_16x16x32_bf16(false, A1.v, false, Bf.v,
                                                    (short)0, a, false, false);
        acc2[t] = a;
    }

    // ---- dz1 = dh1*(1-h1^2); partial g = dz1 @ W1^T ----
    float px[8], py[8];
    #pragma unroll
    for (int rr = 0; rr < 8; ++rr) { px[rr] = 0.0f; py[rr] = 0.0f; }
    #pragma unroll
    for (int t = 0; t < 4; ++t) {
        const int col = t * 16 + r;
        const float w1a = W1[col], w1b = W1[HDIM + col];
        #pragma unroll
        for (int rr = 0; rr < 8; ++rr) {
            const int M = rr + (hi ? 8 : 0);
            const float h1v = (float)h1p[M * HDIM + col];
            const float dz1 = acc2[t][rr] * (1.0f - h1v * h1v);
            px[rr] += dz1 * w1a;
            py[rr] += dz1 * w1b;
        }
    }
    // butterfly reduce across each 16-lane half (columns)
    #pragma unroll
    for (int mask = 1; mask <= 8; mask <<= 1) {
        #pragma unroll
        for (int rr = 0; rr < 8; ++rr) {
            px[rr] += __shfl_xor(px[rr], mask, 32);
            py[rr] += __shfl_xor(py[rr], mask, 32);
        }
    }
    if (r == 0) {
        const int mbase = hi ? 8 : 0;
        #pragma unroll
        for (int rr = 0; rr < 8; ++rr) {
            gp[(mbase + rr) * 2 + 0] = px[rr];
            gp[(mbase + rr) * 2 + 1] = py[rr];
        }
    }
    __syncthreads();

    // ---- scatter: f = -(g . dStrain/dq), lanes 0..15 only ----
    if (lane < 16 && valid) {
        const float gx = gp[r * 2 + 0];   // dE/d eps
        const float gy = gp[r * 2 + 1];   // dE/d kappa

        const float il1 = 1.0f / n1, il2 = 1.0f / n2;
        const float ce  = 0.5f / L_EFF;
        const float ge1x = ce*e1x*il1, ge1y = ce*e1y*il1, ge1z = ce*e1z*il1;
        const float ge2x = ce*e2x*il2, ge2y = ce*e2y*il2, ge2z = ce*e2z*il2;
        const float invD = 1.0f / D;
        const float invk = (kappa > 1e-20f) ? (1.0f / kappa) : 0.0f;
        const float ux = kbx*invk, uy = kby*invk, uz = kbz*invk;
        const float c1 = 2.0f * invD, c2 = kappa * invD;
        const float t1 = n2 * il1,  t2 = n1 * il2;
        const float k1x = c1*(e2y*uz - e2z*uy) - c2*(t1*e1x + e2x);
        const float k1y = c1*(e2z*ux - e2x*uz) - c2*(t1*e1y + e2y);
        const float k1z = c1*(e2x*uy - e2y*ux) - c2*(t1*e1z + e2z);
        const float k2x = c1*(uy*e1z - uz*e1y) - c2*(t2*e2x + e1x);
        const float k2y = c1*(uz*e1x - ux*e1z) - c2*(t2*e2y + e1y);
        const float k2z = c1*(ux*e1y - uy*e1x) - c2*(t2*e2z + e1z);

        const float Gex = gx*ge1x + gy*k1x;   // dE/de1
        const float Gey = gx*ge1y + gy*k1y;
        const float Gez = gx*ge1z + gy*k1z;
        const float Gfx = gx*ge2x + gy*k2x;   // dE/de2
        const float Gfy = gx*ge2y + gy*k2y;
        const float Gfz = gx*ge2z + gy*k2z;

        float* fb = f_el + (long long)bb * ndof;
        atomicAdd(&fb[i0*3+0],  Gex);
        atomicAdd(&fb[i0*3+1],  Gey);
        atomicAdd(&fb[i0*3+2],  Gez);
        atomicAdd(&fb[i1*3+0],  Gfx - Gex);
        atomicAdd(&fb[i1*3+1],  Gfy - Gey);
        atomicAdd(&fb[i1*3+2],  Gfz - Gez);
        atomicAdd(&fb[i2*3+0], -Gfx);
        atomicAdd(&fb[i2*3+1], -Gfy);
        atomicAdd(&fb[i2*3+2], -Gfz);
    }
}

// ---------------------------------------------------------- final assembly --
__global__ void assemble_kernel(float* __restrict__ out,
                                const float* __restrict__ x,
                                const float* __restrict__ f_el,
                                const float* __restrict__ M,
                                const float* __restrict__ C,
                                const float* __restrict__ f_ext,
                                int B, int ndof)
{
    const int n2 = 2 * ndof;
    const int idx = blockIdx.x * blockDim.x + threadIdx.x;
    if (idx >= B * n2) return;
    const int b = idx / n2;
    const int j = idx - b * n2;
    const float* vb = x + (long long)b * n2 + ndof;
    if (j < ndof) {
        out[idx] = vb[j];
    } else {
        const int i = j - ndof;
        const long long di = (long long)i * ndof + i;
        const float rhs = f_el[(long long)b * ndof + i] - C[di] * vb[i] + f_ext[i];
        out[idx] = rhs / M[di];
    }
}

// -----------------------------------------------------------------------------
extern "C" void kernel_launch(void* const* d_in, const int* in_sizes, int n_in,
                              void* d_out, int out_size, void* d_ws, size_t ws_size,
                              hipStream_t stream)
{
    // input order: t, x, springs, W1, b1, W2, b2, W3, b3, M, C, f_ext
    const float* x       = (const float*)d_in[1];
    const int*   springs = (const int*)  d_in[2];
    const float* W1      = (const float*)d_in[3];
    const float* b1      = (const float*)d_in[4];
    const float* W2      = (const float*)d_in[5];
    const float* b2      = (const float*)d_in[6];
    const float* W3      = (const float*)d_in[7];
    const float* M       = (const float*)d_in[9];
    const float* C       = (const float*)d_in[10];
    const float* f_ext   = (const float*)d_in[11];

    const int ndof = in_sizes[11];
    const int B    = in_sizes[1] / (2 * ndof);
    const int S    = in_sizes[2] / 3;

    // ws layout: [0,16KB) packed bf16 W2/W2^T fragments; then f_el (B*ndof f32)
    __bf16* fragbuf = (__bf16*)d_ws;
    float*  f_el    = (float*)((char*)d_ws + 16384);
    float*  out     = (float*)d_out;

    prep_frag_kernel<<<1, 256, 0, stream>>>(W2, fragbuf);

    const int nfe = B * ndof;
    zero_kernel<<<(nfe + 255) / 256, 256, 0, stream>>>(f_el, nfe);

    const long long total = (long long)B * S;
    if (total > 0) {
        const long long tiles  = (total + TILE - 1) / TILE;
        const int       blocks = (int)((tiles + WPB - 1) / WPB);
        spring_force_kernel<<<blocks, 32 * WPB, 0, stream>>>(
            x, springs, W1, b1, b2, W3, fragbuf, f_el, B, S, ndof);
    }

    const int nout = B * 2 * ndof;
    assemble_kernel<<<(nout + 255) / 256, 256, 0, stream>>>(
        out, x, f_el, M, C, f_ext, B, ndof);
}
